// Model_84164179132943
// MI455X (gfx1250) — matrix operand order repaired and checked by reading, not compiled, not run on hardware
//
#include <hip/hip_runtime.h>
#include <hip/hip_bf16.h>

// Problem constants (match reference)
#define NODES 8192
#define EDGES 262144
#define INDIM 512
#define HID   256
// TEMP = 0.5 -> multiply dots by 2.0f

typedef __attribute__((ext_vector_type(16))) __bf16        bf16x16;
typedef __attribute__((ext_vector_type(8)))  float         f32x8;
typedef __attribute__((ext_vector_type(8)))  unsigned short u16x8;

union BF16Tile { bf16x16 v; u16x8 h[2]; };

// ---- WMMA helpers (CDNA5 wave32, 16x16x32 bf16, f32 accum) ----------------

// A-matrix 16x32 bf16 tile (ISA 7.12.2): lane L (0-15) holds row M=L,
// K in {kb..kb+7, kb+16..kb+23} with kb = (L>>4)*8.
__device__ __forceinline__ bf16x16 load_a_tile(const unsigned short* base, int ld) {
    const int lane = threadIdx.x & 31;
    const int r  = lane & 15;
    const int kb = (lane >> 4) << 3;             // 0 or 8
    const unsigned short* p = base + r * ld + kb;
    BF16Tile t;
    t.h[0] = *reinterpret_cast<const u16x8*>(p);
    t.h[1] = *reinterpret_cast<const u16x8*>(p + 16);
    return t.v;
}

// B-matrix 32x16 bf16 tile: lane L (0-15) holds col N=L,
// K in {kb..kb+7, kb+16..kb+23} with kb = (L>>4)*8 (matches A).
__device__ __forceinline__ bf16x16 load_b_tile(const unsigned short* base, int ld) {
    const int lane = threadIdx.x & 31;
    const int c  = lane & 15;
    const int kb = (lane >> 4) << 3;             // 0 or 8
    const unsigned short* p = base + c * ld + kb;
    BF16Tile t;
    t.h[0] = *reinterpret_cast<const u16x8*>(p);
    t.h[1] = *reinterpret_cast<const u16x8*>(p + 16);
    return t.v;
}

__device__ __forceinline__ f32x8 wmma_bf16(bf16x16 a, bf16x16 b, f32x8 c) {
    return __builtin_amdgcn_wmma_f32_16x16x32_bf16(
        /*neg_a=*/false, a, /*neg_b=*/false, b,
        /*c_mod=*/(short)0, c, /*reuse_a=*/false, /*reuse_b=*/false);
}

__device__ __forceinline__ float wave16_sum(float v) {
    v += __shfl_xor(v, 1, 32);
    v += __shfl_xor(v, 2, 32);
    v += __shfl_xor(v, 4, 32);
    v += __shfl_xor(v, 8, 32);
    return v;   // sum within each 16-lane half
}

// ---- Graph prep ------------------------------------------------------------

__global__ void k_degrees(const int* __restrict__ src, const int* __restrict__ dst,
                          float* __restrict__ outDeg, float* __restrict__ inDeg, int E) {
    int e = blockIdx.x * blockDim.x + threadIdx.x;
    if (e >= E) return;
    atomicAdd(outDeg + src[e], 1.0f);
    atomicAdd(inDeg + dst[e], 1.0f);
}

__global__ void k_norms(const float* __restrict__ outDeg, const float* __restrict__ inDeg,
                        float* __restrict__ sn, float* __restrict__ dn, int N) {
    int i = blockIdx.x * blockDim.x + threadIdx.x;
    if (i >= N) return;
    float o = outDeg[i], d = inDeg[i];
    sn[i] = (o > 0.0f) ? rsqrtf(o) : 0.0f;
    dn[i] = (d > 0.0f) ? rsqrtf(d) : 0.0f;
}

// SpMM: agg[dst[e]] += X[src[e]] * (srcScale ? srcScale[src[e]] : 1)
__global__ void k_spmm(const float* __restrict__ X, const int* __restrict__ src,
                       const int* __restrict__ dst, const float* __restrict__ srcScale,
                       float* __restrict__ agg, int E, int D) {
    long long gid = (long long)blockIdx.x * blockDim.x + threadIdx.x;
    int perEdge = D >> 2;                           // float4 chunks per row
    if (gid >= (long long)E * perEdge) return;
    int e = (int)(gid / perEdge);
    int c = (int)(gid % perEdge);
    int s = src[e], d = dst[e];
    float4 x = reinterpret_cast<const float4*>(X + (size_t)s * D)[c];
    float sc = srcScale ? srcScale[s] : 1.0f;
    float* out = agg + (size_t)d * D + (size_t)c * 4;
    atomicAdd(out + 0, x.x * sc);
    atomicAdd(out + 1, x.y * sc);
    atomicAdd(out + 2, x.z * sc);
    atomicAdd(out + 3, x.w * sc);
}

// Y_bf16[i] = bf16( X[i] * (rowScale ? rowScale[i/D] : 1) )
__global__ void k_scale_cvt(const float* __restrict__ X, const float* __restrict__ rowScale,
                            __hip_bfloat16* __restrict__ Y, int total, int D) {
    int i = blockIdx.x * blockDim.x + threadIdx.x;
    if (i >= total) return;
    float s = rowScale ? rowScale[i / D] : 1.0f;
    Y[i] = __float2bfloat16(X[i] * s);
}

// Wt[n*K + k] = bf16( W[k*Nn + n] )  (pre-transpose weights for B-tile loads)
__global__ void k_transpose_cvt(const float* __restrict__ W, __hip_bfloat16* __restrict__ Wt,
                                int K, int Nn) {
    int i = blockIdx.x * blockDim.x + threadIdx.x;
    if (i >= K * Nn) return;
    int n = i / K, k = i % K;
    Wt[i] = __float2bfloat16(W[k * Nn + n]);
}

// ---- Dense GEMM: C[MxN] = A[MxK](bf16) * Bt[NxK](bf16)^T + bias; opt relu, row scale
__global__ __launch_bounds__(256) void k_gemm_bf16(
    const unsigned short* __restrict__ A, const unsigned short* __restrict__ Bt,
    const float* __restrict__ bias, float* __restrict__ C,
    const float* __restrict__ rowScale, int M, int N, int K, int relu, int ntPerSplit) {
    const int wave = threadIdx.x >> 5;
    const int lane = threadIdx.x & 31;
    const int m0 = blockIdx.x * 128 + wave * 16;
    if (m0 >= M) return;
    const int nt0 = blockIdx.y * ntPerSplit;
    const int ksteps = K >> 5;
    for (int nt = nt0; nt < nt0 + ntPerSplit; ++nt) {
        f32x8 acc = {0.f, 0.f, 0.f, 0.f, 0.f, 0.f, 0.f, 0.f};
        const unsigned short* arow = A + (size_t)m0 * K;
        const unsigned short* brow = Bt + (size_t)nt * 16 * K;
        for (int kt = 0; kt < ksteps; ++kt) {
            bf16x16 a = load_a_tile(arow + kt * 32, K);
            bf16x16 b = load_b_tile(brow + kt * 32, K);
            acc = wmma_bf16(a, b, acc);
        }
        const int col   = nt * 16 + (lane & 15);
        const int rbase = m0 + ((lane >> 4) << 3);
        const float bv = bias ? bias[col] : 0.0f;
#pragma unroll
        for (int i = 0; i < 8; ++i) {
            int row = rbase + i;
            float v = acc[i] + bv;
            if (relu) v = fmaxf(v, 0.0f);
            if (rowScale) v *= rowScale[row];
            C[(size_t)row * N + col] = v;
        }
    }
}

// ---- negSim[i] += sum_j exp(2 * z_i . z_j), LDS-staged panels of z ---------
#define NEG_PANEL 128
__global__ __launch_bounds__(256) void k_negsim(
    const unsigned short* __restrict__ zbf, float* __restrict__ negSim,
    int N, int colsPerSplit) {
    __shared__ unsigned short panel[NEG_PANEL * HID];   // 64 KB of 320 KB/WGP
    const int wave = threadIdx.x >> 5;
    const int lane = threadIdx.x & 31;
    const int m0 = blockIdx.x * 128 + wave * 16;

    bf16x16 a[8];                                       // full K=256 of 16 rows in VGPRs
#pragma unroll
    for (int kt = 0; kt < 8; ++kt)
        a[kt] = load_a_tile(zbf + (size_t)m0 * HID + kt * 32, HID);

    float rs[8] = {0, 0, 0, 0, 0, 0, 0, 0};
    const int c0beg = blockIdx.y * colsPerSplit;
    for (int c0 = c0beg; c0 < c0beg + colsPerSplit; c0 += NEG_PANEL) {
        const uint4* gsrc = reinterpret_cast<const uint4*>(zbf + (size_t)c0 * HID);
        uint4* dst4 = reinterpret_cast<uint4*>(panel);
        for (int i = threadIdx.x; i < NEG_PANEL * HID / 8; i += 256)
            dst4[i] = gsrc[i];
        if (c0 + NEG_PANEL < c0beg + colsPerSplit)
            __builtin_prefetch(zbf + (size_t)(c0 + NEG_PANEL) * HID, 0, 1);
        __syncthreads();
#pragma unroll
        for (int nt = 0; nt < NEG_PANEL / 16; ++nt) {
            f32x8 acc = {0.f, 0.f, 0.f, 0.f, 0.f, 0.f, 0.f, 0.f};
#pragma unroll
            for (int kt = 0; kt < 8; ++kt) {
                bf16x16 b = load_b_tile(panel + nt * 16 * HID + kt * 32, HID);
                acc = wmma_bf16(a[kt], b, acc);
            }
#pragma unroll
            for (int i = 0; i < 8; ++i)
                rs[i] += __expf(acc[i] * 2.0f);          // /TEMP == *2
        }
        __syncthreads();
    }
#pragma unroll
    for (int i = 0; i < 8; ++i) rs[i] = wave16_sum(rs[i]);
    if ((lane & 15) == 0) {
        const int rbase = m0 + ((lane >> 4) << 3);
#pragma unroll
        for (int i = 0; i < 8; ++i) atomicAdd(&negSim[rbase + i], rs[i]);
    }
}

// ---- row-wise L2 normalize (warp per row, D=256) ---------------------------
__global__ __launch_bounds__(256) void k_l2norm(
    const float* __restrict__ X, float* __restrict__ Yf,
    __hip_bfloat16* __restrict__ Ybf, int N, int D) {
    const int warp = (blockIdx.x * blockDim.x + threadIdx.x) >> 5;
    if (warp >= N) return;
    const int lane = threadIdx.x & 31;
    const float4* row = reinterpret_cast<const float4*>(X + (size_t)warp * D);
    float4 x0 = row[lane * 2], x1 = row[lane * 2 + 1];
    float s = x0.x * x0.x + x0.y * x0.y + x0.z * x0.z + x0.w * x0.w +
              x1.x * x1.x + x1.y * x1.y + x1.z * x1.z + x1.w * x1.w;
    s += __shfl_xor(s, 1, 32);  s += __shfl_xor(s, 2, 32);
    s += __shfl_xor(s, 4, 32);  s += __shfl_xor(s, 8, 32);
    s += __shfl_xor(s, 16, 32);
    float scale = 1.0f / fmaxf(sqrtf(s), 1e-12f);
    x0.x *= scale; x0.y *= scale; x0.z *= scale; x0.w *= scale;
    x1.x *= scale; x1.y *= scale; x1.z *= scale; x1.w *= scale;
    if (Yf) {
        float4* orow = reinterpret_cast<float4*>(Yf + (size_t)warp * D);
        orow[lane * 2] = x0; orow[lane * 2 + 1] = x1;
    }
    if (Ybf) {
        __hip_bfloat16* ob = Ybf + (size_t)warp * D + lane * 8;
        ob[0] = __float2bfloat16(x0.x); ob[1] = __float2bfloat16(x0.y);
        ob[2] = __float2bfloat16(x0.z); ob[3] = __float2bfloat16(x0.w);
        ob[4] = __float2bfloat16(x1.x); ob[5] = __float2bfloat16(x1.y);
        ob[6] = __float2bfloat16(x1.z); ob[7] = __float2bfloat16(x1.w);
    }
}

// ---- per-edge scores: sim -> posSum[dst]; log(negSim[dst]+exp(sim)) -> negSum[dst]
__global__ __launch_bounds__(256) void k_edge_scores(
    const float* __restrict__ un, const float* __restrict__ q,
    const int* __restrict__ src, const int* __restrict__ dst,
    const float* __restrict__ negSim, float* __restrict__ posSum,
    float* __restrict__ negSum, int E) {
    const int warp = (blockIdx.x * blockDim.x + threadIdx.x) >> 5;
    if (warp >= E) return;
    const int lane = threadIdx.x & 31;
    const int s = src[warp], d = dst[warp];
    const float4* ur = reinterpret_cast<const float4*>(un + (size_t)s * HID);
    const float4* qr = reinterpret_cast<const float4*>(q  + (size_t)d * HID);
    float4 a0 = ur[lane * 2], a1 = ur[lane * 2 + 1];
    float4 b0 = qr[lane * 2], b1 = qr[lane * 2 + 1];
    float dot = a0.x * b0.x + a0.y * b0.y + a0.z * b0.z + a0.w * b0.w +
                a1.x * b1.x + a1.y * b1.y + a1.z * b1.z + a1.w * b1.w;
    dot += __shfl_xor(dot, 1, 32);  dot += __shfl_xor(dot, 2, 32);
    dot += __shfl_xor(dot, 4, 32);  dot += __shfl_xor(dot, 8, 32);
    dot += __shfl_xor(dot, 16, 32);
    if (lane == 0) {
        float sim = dot * 2.0f;                          // / TEMP
        atomicAdd(posSum + d, sim);
        float m = __logf(negSim[d] + __expf(sim));
        atomicAdd(negSum + d, m);
    }
}

// ---- final scalar: mean_i( in_deg>0 ? (negSum-posSum)/in_deg : 0 ) ---------
__global__ __launch_bounds__(256) void k_final(
    const float* __restrict__ posSum, const float* __restrict__ negSum,
    const float* __restrict__ inDeg, float* __restrict__ out, int N) {
    __shared__ float red[256];
    float s = 0.0f;
    for (int i = threadIdx.x; i < N; i += 256) {
        float deg = inDeg[i];
        if (deg > 0.0f) s += (negSum[i] - posSum[i]) / deg;
    }
    red[threadIdx.x] = s;
    __syncthreads();
    for (int w = 128; w > 0; w >>= 1) {
        if (threadIdx.x < w) red[threadIdx.x] += red[threadIdx.x + w];
        __syncthreads();
    }
    if (threadIdx.x == 0) out[0] = red[0] / (float)N;
}

// ---------------------------------------------------------------------------

extern "C" void kernel_launch(void* const* d_in, const int* in_sizes, int n_in,
                              void* d_out, int out_size, void* d_ws, size_t ws_size,
                              hipStream_t stream) {
    (void)in_sizes; (void)n_in; (void)out_size; (void)ws_size;
    const int N = NODES, E = EDGES;

    const float* feat = (const float*)d_in[0];
    const int*   src  = (const int*)  d_in[1];
    const int*   dst  = (const int*)  d_in[2];
    const float* W1  = (const float*)d_in[3];  const float* b1  = (const float*)d_in[4];
    const float* W2  = (const float*)d_in[5];  const float* b2  = (const float*)d_in[6];
    const float* Wt1 = (const float*)d_in[7];  const float* bt1 = (const float*)d_in[8];
    const float* Wt2 = (const float*)d_in[9];  const float* bt2 = (const float*)d_in[10];
    const float* Wp  = (const float*)d_in[11]; const float* bp  = (const float*)d_in[12];

    char* ws = (char*)d_ws;
    size_t off = 0;
    auto alloc = [&](size_t bytes) -> char* {
        char* p = ws + off; off += (bytes + 255) & ~(size_t)255; return p;
    };

    float* outDeg  = (float*)alloc((size_t)N * 4);
    float* inDeg   = (float*)alloc((size_t)N * 4);
    float* srcNorm = (float*)alloc((size_t)N * 4);
    float* dstNorm = (float*)alloc((size_t)N * 4);
    float* agg1    = (float*)alloc((size_t)N * INDIM * 4);      // reused: agg2v | agg2u
    __hip_bfloat16* A1bf = (__hip_bfloat16*)alloc((size_t)N * INDIM * 2); // reused: A2v|A2u then vbf|zbf
    float* hsv     = (float*)alloc((size_t)N * HID * 4);        // reused: v
    float* hsu     = (float*)alloc((size_t)N * HID * 4);        // reused: u -> un (in place)
    float* p       = (float*)alloc((size_t)N * HID * 4);        // -> q (in place)
    __hip_bfloat16* W1t  = (__hip_bfloat16*)alloc((size_t)INDIM * HID * 2);
    __hip_bfloat16* Wt1t = (__hip_bfloat16*)alloc((size_t)INDIM * HID * 2);
    __hip_bfloat16* W2t  = (__hip_bfloat16*)alloc((size_t)HID * HID * 2);
    __hip_bfloat16* Wt2t = (__hip_bfloat16*)alloc((size_t)HID * HID * 2);
    __hip_bfloat16* Wpt  = (__hip_bfloat16*)alloc((size_t)HID * HID * 2);
    float* posSum  = (float*)alloc((size_t)N * 4);
    float* negSum  = (float*)alloc((size_t)N * 4);
    float* negSim  = (float*)alloc((size_t)N * 4);

    // Aliased views (lifetimes are disjoint, stream-ordered)
    float* agg2v = agg1;
    float* agg2u = agg1 + (size_t)N * HID;
    __hip_bfloat16* A2vbf = A1bf;
    __hip_bfloat16* A2ubf = A1bf + (size_t)N * HID;
    __hip_bfloat16* vbf   = A1bf;                       // after GEMM2v done
    __hip_bfloat16* zbf   = A1bf + (size_t)N * HID;     // after GEMM2u done
    float* v = hsv;  // after SpMM2 done
    float* u = hsu;

    // --- degrees & norms ---
    hipMemsetAsync(outDeg, 0, (size_t)N * 4, stream);
    hipMemsetAsync(inDeg,  0, (size_t)N * 4, stream);
    k_degrees<<<(E + 255) / 256, 256, 0, stream>>>(src, dst, outDeg, inDeg, E);
    k_norms<<<(N + 255) / 256, 256, 0, stream>>>(outDeg, inDeg, srcNorm, dstNorm, N);

    // --- weight prep (transpose + bf16) ---
    k_transpose_cvt<<<(INDIM * HID + 255) / 256, 256, 0, stream>>>(W1,  W1t,  INDIM, HID);
    k_transpose_cvt<<<(INDIM * HID + 255) / 256, 256, 0, stream>>>(Wt1, Wt1t, INDIM, HID);
    k_transpose_cvt<<<(HID * HID + 255) / 256, 256, 0, stream>>>(W2,  W2t,  HID, HID);
    k_transpose_cvt<<<(HID * HID + 255) / 256, 256, 0, stream>>>(Wt2, Wt2t, HID, HID);
    k_transpose_cvt<<<(HID * HID + 255) / 256, 256, 0, stream>>>(Wp,  Wpt,  HID, HID);

    // --- layer 1: shared aggregation (both encoders use feat * src_norm) ---
    hipMemsetAsync(agg1, 0, (size_t)N * INDIM * 4, stream);
    {
        long long tot = (long long)E * (INDIM / 4);
        k_spmm<<<(unsigned)((tot + 255) / 256), 256, 0, stream>>>(feat, src, dst, srcNorm, agg1, E, INDIM);
    }
    k_scale_cvt<<<(N * INDIM + 255) / 256, 256, 0, stream>>>(agg1, dstNorm, A1bf, N * INDIM, INDIM);

    dim3 gGemm(N / 128, 4);
    // h = relu(agg1n @ W + b) * src_norm  (src_norm pre-applied for layer-2 SpMM)
    k_gemm_bf16<<<gGemm, 256, 0, stream>>>((const unsigned short*)A1bf, (const unsigned short*)W1t,
                                           b1, hsv, srcNorm, N, HID, INDIM, 1, (HID / 16) / 4);
    k_gemm_bf16<<<gGemm, 256, 0, stream>>>((const unsigned short*)A1bf, (const unsigned short*)Wt1t,
                                           bt1, hsu, srcNorm, N, HID, INDIM, 1, (HID / 16) / 4);

    // --- layer 2 SpMMs ---
    hipMemsetAsync(agg2v, 0, (size_t)N * HID * 4 * 2, stream);
    {
        long long tot = (long long)E * (HID / 4);
        k_spmm<<<(unsigned)((tot + 255) / 256), 256, 0, stream>>>(hsv, src, dst, nullptr, agg2v, E, HID);
        k_spmm<<<(unsigned)((tot + 255) / 256), 256, 0, stream>>>(hsu, src, dst, nullptr, agg2u, E, HID);
    }
    k_scale_cvt<<<(N * HID + 255) / 256, 256, 0, stream>>>(agg2v, dstNorm, A2vbf, N * HID, HID);
    k_scale_cvt<<<(N * HID + 255) / 256, 256, 0, stream>>>(agg2u, dstNorm, A2ubf, N * HID, HID);

    // --- layer 2 GEMMs: v, u (overwrite hsv/hsu) ---
    k_gemm_bf16<<<gGemm, 256, 0, stream>>>((const unsigned short*)A2vbf, (const unsigned short*)W2t,
                                           b2, v, nullptr, N, HID, HID, 0, (HID / 16) / 4);
    k_gemm_bf16<<<gGemm, 256, 0, stream>>>((const unsigned short*)A2ubf, (const unsigned short*)Wt2t,
                                           bt2, u, nullptr, N, HID, HID, 0, (HID / 16) / 4);

    // --- projector: p = v @ Wp + bp ---
    k_scale_cvt<<<(N * HID + 255) / 256, 256, 0, stream>>>(v, nullptr, vbf, N * HID, HID);
    k_gemm_bf16<<<gGemm, 256, 0, stream>>>((const unsigned short*)vbf, (const unsigned short*)Wpt,
                                           bp, p, nullptr, N, HID, HID, 0, (HID / 16) / 4);

    // --- normalizations: q (in-place p), un (in-place u), z -> bf16 ---
    k_l2norm<<<N / 8, 256, 0, stream>>>(p, p, nullptr, N, HID);   // q
    k_l2norm<<<N / 8, 256, 0, stream>>>(u, u, nullptr, N, HID);   // un
    k_l2norm<<<N / 8, 256, 0, stream>>>(v, nullptr, zbf, N, HID); // z (bf16 for WMMA)

    // --- neg_sim: row sums of exp(z z^T / TEMP), WMMA + LDS panels ---
    hipMemsetAsync(negSim, 0, (size_t)N * 4, stream);
    dim3 gNeg(N / 128, 4);
    k_negsim<<<gNeg, 256, 0, stream>>>((const unsigned short*)zbf, negSim, N, N / 4);

    // --- per-edge scores + segment sums ---
    hipMemsetAsync(posSum, 0, (size_t)N * 4, stream);
    hipMemsetAsync(negSum, 0, (size_t)N * 4, stream);
    k_edge_scores<<<E / 8, 256, 0, stream>>>(u, p, src, dst, negSim, posSum, negSum, E);

    // --- final scalar ---
    k_final<<<1, 256, 0, stream>>>(posSum, negSum, inDeg, (float*)d_out, N);
}
